// SpatioTemporalModel_15822659518640
// MI455X (gfx1250) — compile-verified, matrix operand order checked by reading, not run on hardware
//
#include <hip/hip_runtime.h>

typedef __attribute__((ext_vector_type(16))) _Float16 v16h;
typedef __attribute__((ext_vector_type(8)))  _Float16 v8h;
typedef __attribute__((ext_vector_type(8)))  float    v8f;
typedef __attribute__((ext_vector_type(4)))  float    v4f;

#define N_NODES 50000
#define N_EDGES 800000
#define BATCH 4
#define SEQ 96
#define FEAT 28
#define HID 128
#define HEADS 8
#define HD 16
#define PRED 24

// ---------------------------------------------------------------- helpers

__device__ __forceinline__ void atomicMaxF(float* addr, float val) {
  if (val >= 0.0f) atomicMax((int*)addr, __float_as_int(val));
  else             atomicMin((unsigned int*)addr, __float_as_uint(val));
}

__global__ void fill_kernel(float* p, float v, int n) {
  int i = blockIdx.x * blockDim.x + threadIdx.x;
  if (i < n) p[i] = v;
}

// Zero-pad A rows from Korig to Kpad (row-major).
__global__ void pad_a_kernel(const float* __restrict__ A, float* __restrict__ Ap,
                             int Korig, int Kpad, int total)
{
  int i = blockIdx.x * blockDim.x + threadIdx.x;
  if (i >= total) return;
  int r = i / Kpad, k = i - r * Kpad;
  Ap[i] = (k < Korig) ? A[(size_t)r * Korig + k] : 0.0f;
}

// Transpose+convert weights: Bt[n][k] (f16, K zero-padded) from B[k][n] (f32, row stride ldb).
__global__ void pack_b_kernel(const float* __restrict__ B, _Float16* __restrict__ Bt,
                              int Korig, int Kpad, int N, int ldb)
{
  int i = blockIdx.x * blockDim.x + threadIdx.x;
  if (i >= N * Kpad) return;
  int n = i / Kpad, k = i - n * Kpad;
  Bt[i] = (k < Korig) ? (_Float16)B[(size_t)k * ldb + n] : (_Float16)0.0f;
}

// ---------------------------------------------------------------- WMMA GEMM
// C[M,N] = act(A[M,K] @ B + bias). K % 32 == 0, N % 32 == 0.
// A: f32 row-major [M,K]. Bt: f16 [N,K] (pre-transposed/padded).
// One wave computes a 16x32 strip (two 16x16 WMMA tiles sharing the A
// operand); 4 waves (4 consecutive M tiles) per block.
__global__ __launch_bounds__(128) void gemm_wmma(
    const float* __restrict__ A, const _Float16* __restrict__ Bt,
    const float* __restrict__ bias, float* __restrict__ C,
    int M, int N, int K, int act)
{
  int wave = threadIdx.x >> 5, lane = threadIdx.x & 31;
  int mt = blockIdx.x * 4 + wave;
  if (mt * 16 >= M) return;            // wave-uniform; no barriers below
  int nt = blockIdx.y;                 // 32-column strip index
  int col = lane & 15, hi = lane >> 4;
  const float*    ap  = A  + (size_t)(mt * 16 + col) * K + hi * 8;
  const _Float16* bp0 = Bt + (size_t)(nt * 32 + col) * K + hi * 8;
  const _Float16* bp1 = bp0 + (size_t)16 * K;
  int n0 = nt * 32 + col;
  v8f acc0 = {}, acc1 = {};
  for (int k0 = 0; k0 < K; k0 += 32) {
    __builtin_prefetch(ap + k0 + 64, 0, 1);   // next A chunk -> global_prefetch
    v4f a0 = *(const v4f*)(ap + k0);
    v4f a1 = *(const v4f*)(ap + k0 + 4);
    v4f a2 = *(const v4f*)(ap + k0 + 16);
    v4f a3 = *(const v4f*)(ap + k0 + 20);
    v16h a;
#pragma unroll
    for (int j = 0; j < 4; ++j) {
      a[j]      = (_Float16)a0[j];
      a[4 + j]  = (_Float16)a1[j];
      a[8 + j]  = (_Float16)a2[j];
      a[12 + j] = (_Float16)a3[j];
    }
    v8h b00 = *(const v8h*)(bp0 + k0);
    v8h b01 = *(const v8h*)(bp0 + k0 + 16);
    v16h b0 = __builtin_shufflevector(b00, b01, 0, 1, 2, 3, 4, 5, 6, 7,
                                      8, 9, 10, 11, 12, 13, 14, 15);
    v8h b10 = *(const v8h*)(bp1 + k0);
    v8h b11 = *(const v8h*)(bp1 + k0 + 16);
    v16h b1 = __builtin_shufflevector(b10, b11, 0, 1, 2, 3, 4, 5, 6, 7,
                                      8, 9, 10, 11, 12, 13, 14, 15);
    acc0 = __builtin_amdgcn_wmma_f32_16x16x32_f16(false, a, false, b0,
                                                  (short)0, acc0, false, false);
    acc1 = __builtin_amdgcn_wmma_f32_16x16x32_f16(false, a, false, b1,
                                                  (short)0, acc1, false, false);
  }
  float bb0 = bias ? bias[n0] : 0.0f;
  float bb1 = bias ? bias[n0 + 16] : 0.0f;
#pragma unroll
  for (int v = 0; v < 8; ++v) {
    float c0 = acc0[v] + bb0;
    float c1 = acc1[v] + bb1;
    if (act == 1) { c0 = fmaxf(c0, 0.0f); c1 = fmaxf(c1, 0.0f); }
    size_t r = (size_t)(mt * 16 + v + 8 * hi) * N;
    C[r + n0]      = c0;
    C[r + n0 + 16] = c1;
  }
}

// ---------------------------------------------------------------- GAT pieces

__global__ void gat_alpha_kernel(const float* __restrict__ H,
                                 const float* __restrict__ a_src,
                                 const float* __restrict__ a_dst,
                                 float* __restrict__ as_, float* __restrict__ ad_,
                                 int NH8)
{
  int i = blockIdx.x * blockDim.x + threadIdx.x;
  if (i >= NH8) return;
  int nn = i >> 3, h = i & 7;
  const float* hr = H + (size_t)nn * HID + h * HD;
  float s = 0.f, d = 0.f;
#pragma unroll
  for (int k = 0; k < HD; ++k) {
    s += hr[k] * a_src[h * HD + k];
    d += hr[k] * a_dst[h * HD + k];
  }
  as_[i] = s; ad_[i] = d;
}

__global__ void edge_max_kernel(const int* __restrict__ src, const int* __restrict__ dst,
                                const float* __restrict__ asrc, const float* __restrict__ adst,
                                float* __restrict__ mmax, int EH)
{
  int i = blockIdx.x * blockDim.x + threadIdx.x;
  if (i >= EH) return;
  int e = i >> 3, h = i & 7;
  int s = src[e], d = dst[e];
  float v = asrc[s * HEADS + h] + adst[d * HEADS + h];
  v = v > 0.f ? v : 0.2f * v;
  atomicMaxF(&mmax[d * HEADS + h], v);
}

__global__ void edge_expsum_kernel(const int* __restrict__ src, const int* __restrict__ dst,
                                   const float* __restrict__ asrc, const float* __restrict__ adst,
                                   const float* __restrict__ mmax,
                                   float* __restrict__ exb, float* __restrict__ ssum, int EH)
{
  int i = blockIdx.x * blockDim.x + threadIdx.x;
  if (i >= EH) return;
  int e = i >> 3, h = i & 7;
  int s = src[e], d = dst[e];
  float v = asrc[s * HEADS + h] + adst[d * HEADS + h];
  v = v > 0.f ? v : 0.2f * v;
  float ex = __expf(v - mmax[d * HEADS + h]);
  exb[i] = ex;
  atomicAdd(&ssum[d * HEADS + h], ex);
}

__global__ void edge_agg_kernel(const int* __restrict__ src, const int* __restrict__ dst,
                                const float* __restrict__ exb, const float* __restrict__ ssum,
                                const float* __restrict__ H, float* __restrict__ agg, int EH)
{
  int i = blockIdx.x * blockDim.x + threadIdx.x;
  if (i >= EH) return;
  int e = i >> 3, h = i & 7;
  int s = src[e], d = dst[e];
  float w = exb[i] / (ssum[d * HEADS + h] + 1e-16f);
  const float* hr = H + (size_t)s * HID + h * HD;
  float* ar = agg + (size_t)d * HID + h * HD;
#pragma unroll
  for (int k = 0; k < HD; ++k) atomicAdd(&ar[k], hr[k] * w);
}

// out[row] = LN( (elu? elu(a) : a) + res ) * g + b   (128 features per row)
__global__ __launch_bounds__(128) void residual_ln_kernel(
    const float* __restrict__ a, const float* __restrict__ res,
    const float* __restrict__ g, const float* __restrict__ bta,
    float* __restrict__ out, int do_elu)
{
  __shared__ float sh[HID];
  int row = blockIdx.x, d = threadIdx.x;
  float v = a[(size_t)row * HID + d];
  if (do_elu) v = v > 0.f ? v : (__expf(v) - 1.f);
  if (res) v += res[(size_t)row * HID + d];
  sh[d] = v; __syncthreads();
  for (int off = 64; off > 0; off >>= 1) { if (d < off) sh[d] += sh[d + off]; __syncthreads(); }
  float mu = sh[0] * (1.0f / HID); __syncthreads();
  float c = v - mu;
  sh[d] = c * c; __syncthreads();
  for (int off = 64; off > 0; off >>= 1) { if (d < off) sh[d] += sh[d + off]; __syncthreads(); }
  float var = sh[0] * (1.0f / HID);
  out[(size_t)row * HID + d] = c * rsqrtf(var + 1e-5f) * g[d] + bta[d];
}

// ---------------------------------------------------------------- temporal

__global__ void add_pe_kernel(float* t, int n)
{
  int i = blockIdx.x * blockDim.x + threadIdx.x;
  if (i >= n) return;
  int d = i % HID;
  int s = (i / HID) % SEQ;
  float i2 = (float)(d & ~1);
  float ang = (float)s * __powf(10000.0f, -i2 / (float)HID);
  t[i] += (d & 1) ? __cosf(ang) : __sinf(ang);
}

// qkv: [B*S, 384] with q|k|v column blocks; out: [B*S, 128]
__global__ __launch_bounds__(96) void temporal_attn_kernel(
    const float* __restrict__ qkv, float* __restrict__ out)
{
  __shared__ float sc[SEQ * SEQ];
  int b = blockIdx.x >> 3, h = blockIdx.x & 7;
  int q = threadIdx.x;
  const float* qrow = qkv + (size_t)(b * SEQ + q) * (3 * HID) + h * HD;
  float mx = -1e30f;
  for (int k = 0; k < SEQ; ++k) {
    const float* krow = qkv + (size_t)(b * SEQ + k) * (3 * HID) + HID + h * HD;
    float s = 0.f;
#pragma unroll
    for (int d = 0; d < HD; ++d) s += qrow[d] * krow[d];
    s *= 0.25f;                 // 1/sqrt(16)
    sc[q * SEQ + k] = s;
    mx = fmaxf(mx, s);
  }
  float sum = 0.f;
  for (int k = 0; k < SEQ; ++k) { float e = __expf(sc[q * SEQ + k] - mx); sc[q * SEQ + k] = e; sum += e; }
  float inv = 1.0f / sum;
  for (int d = 0; d < HD; ++d) {
    float o = 0.f;
    for (int k = 0; k < SEQ; ++k)
      o += sc[q * SEQ + k] * qkv[(size_t)(b * SEQ + k) * (3 * HID) + 2 * HID + h * HD + d];
    out[(size_t)(b * SEQ + q) * HID + h * HD + d] = o * inv;
  }
}

// ---------------------------------------------------------------- fusion attention
// Every node queries the 96-step temporal sequence. One wave per
// (16-node tile, batch, head): 6 QK^T WMMAs (d=16 zero-padded to K=32) ->
// softmax via half-wave shfl reductions -> restage P in LDS (A-operand
// layout) -> 3 PV WMMAs.
__global__ __launch_bounds__(128) void fusion_attn_kernel(
    const float* __restrict__ qp, const float* __restrict__ kp,
    const float* __restrict__ vp, float* __restrict__ outH)
{
  __shared__ _Float16 pbuf[4][16 * SEQ];
  int wave = threadIdx.x >> 5, lane = threadIdx.x & 31;
  int col = lane & 15, hi = lane >> 4;
  int mt = blockIdx.x * 4 + wave;
  int b = blockIdx.y >> 3, h = blockIdx.y & 7;
  bool active = (mt * 16 < N_NODES);

  v8f acc[6] = {};
  if (active) {
    // Q operand: slots 0..7 hold d = hi*8 + s; slots 8..15 are structural zeros.
    const float* qrow = qp + (size_t)(mt * 16 + col) * HID + h * HD + hi * 8;
    v4f q0 = *(const v4f*)qrow;
    v4f q1 = *(const v4f*)(qrow + 4);
    v16h qa;
#pragma unroll
    for (int j = 0; j < 4; ++j) {
      qa[j]      = (_Float16)q0[j];
      qa[4 + j]  = (_Float16)q1[j];
      qa[8 + j]  = (_Float16)0.0f;
      qa[12 + j] = (_Float16)0.0f;
    }
#pragma unroll
    for (int t = 0; t < 6; ++t) {
      const float* krow = kp + (size_t)(b * SEQ + t * 16 + col) * HID + h * HD + hi * 8;
      v4f k0v = *(const v4f*)krow;
      v4f k1v = *(const v4f*)(krow + 4);
      v16h kb;
#pragma unroll
      for (int j = 0; j < 4; ++j) {
        kb[j]      = (_Float16)k0v[j];
        kb[4 + j]  = (_Float16)k1v[j];
        kb[8 + j]  = (_Float16)0.0f;
        kb[12 + j] = (_Float16)0.0f;
      }
      acc[t] = __builtin_amdgcn_wmma_f32_16x16x32_f16(false, qa, false, kb,
                                                      (short)0, acc[t], false, false);
    }
    // rowwise softmax over 96: each row's 16 cols of a tile live in one
    // 16-lane half of one accumulator VGPR -> xor-shuffle within half.
    float lm[8], ls[8];
#pragma unroll
    for (int v = 0; v < 8; ++v) {
      float m_ = -1e30f;
#pragma unroll
      for (int t = 0; t < 6; ++t) m_ = fmaxf(m_, acc[t][v] * 0.25f);
      lm[v] = m_;
    }
#pragma unroll
    for (int off = 1; off < 16; off <<= 1)
#pragma unroll
      for (int v = 0; v < 8; ++v) lm[v] = fmaxf(lm[v], __shfl_xor(lm[v], off, 32));
#pragma unroll
    for (int v = 0; v < 8; ++v) ls[v] = 0.f;
#pragma unroll
    for (int t = 0; t < 6; ++t)
#pragma unroll
      for (int v = 0; v < 8; ++v) {
        float p = __expf(acc[t][v] * 0.25f - lm[v]);
        acc[t][v] = p;
        ls[v] += p;
      }
#pragma unroll
    for (int off = 1; off < 16; off <<= 1)
#pragma unroll
      for (int v = 0; v < 8; ++v) ls[v] += __shfl_xor(ls[v], off, 32);
#pragma unroll
    for (int t = 0; t < 6; ++t)
#pragma unroll
      for (int v = 0; v < 8; ++v)
        pbuf[wave][(v + 8 * hi) * SEQ + t * 16 + col] =
            (_Float16)(acc[t][v] / (ls[v] + 1e-16f));
  }
  __syncthreads();
  if (active) {
    v8f oacc = {};
#pragma unroll
    for (int k0 = 0; k0 < SEQ; k0 += 32) {
      // P operand from LDS: contiguous 8-element runs -> two b128 ds loads.
      const _Float16* prow = &pbuf[wave][col * SEQ + k0 + hi * 8];
      v8h p0 = *(const v8h*)prow;
      v8h p1 = *(const v8h*)(prow + 16);
      v16h pa = __builtin_shufflevector(p0, p1, 0, 1, 2, 3, 4, 5, 6, 7,
                                        8, 9, 10, 11, 12, 13, 14, 15);
      v16h vb;
#pragma unroll
      for (int s = 0; s < 16; ++s) {
        int kk = k0 + hi * 8 + ((s >> 3) << 4) + (s & 7);
        vb[s] = (_Float16)vp[(size_t)(b * SEQ + kk) * HID + h * HD + col];
      }
      oacc = __builtin_amdgcn_wmma_f32_16x16x32_f16(false, pa, false, vb,
                                                    (short)0, oacc, false, false);
    }
#pragma unroll
    for (int v = 0; v < 8; ++v)
      outH[((size_t)b * N_NODES + mt * 16 + v + 8 * hi) * HID + h * HD + col] = oacc[v];
  }
}

// ---------------------------------------------------------------- output head
// out = h1[64] . o2_w + o2_b ; ph = relu(out*h1_w + h1_b) ; pred = ph @ h2_w + h2_b
__global__ void head_kernel(const float* __restrict__ h1,
                            const float* __restrict__ o2w, const float* __restrict__ o2b,
                            const float* __restrict__ h1w, const float* __restrict__ h1b,
                            const float* __restrict__ h2w, const float* __restrict__ h2b,
                            float* __restrict__ out, int rows)
{
  int r = blockIdx.x * blockDim.x + threadIdx.x;
  if (r >= rows) return;
  const float* hr = h1 + (size_t)r * 64;
  float o = o2b[0];
#pragma unroll
  for (int j = 0; j < 64; ++j) o += hr[j] * o2w[j];
  float ph[8];
#pragma unroll
  for (int j = 0; j < 8; ++j) { float v = o * h1w[j] + h1b[j]; ph[j] = v > 0.f ? v : 0.f; }
  float* pr = out + (size_t)r * PRED;
#pragma unroll
  for (int k = 0; k < PRED; ++k) {
    float p = h2b[k];
#pragma unroll
    for (int j = 0; j < 8; ++j) p += ph[j] * h2w[j * PRED + k];
    pr[k] = p;
  }
}

// ---------------------------------------------------------------- launch

extern "C" void kernel_launch(void* const* d_in, const int* in_sizes, int n_in,
                              void* d_out, int out_size, void* d_ws, size_t ws_size,
                              hipStream_t stream)
{
  (void)in_sizes; (void)n_in; (void)out_size; (void)ws_size;
  const float* nf = (const float*)d_in[0];
  const int*   ei = (const int*)d_in[1];
  const float* ts = (const float*)d_in[2];
  const int* src = ei;
  const int* dst = ei + N_EDGES;
  auto P = [&](int i) { return (const float*)d_in[i]; };

  // param indices (setup_inputs insertion order)
  const int SP_IN_W = 3, SP_IN_B = 4;
  const int GAT0 = 5;                 // per layer: W, a_src, a_dst, ln_g, ln_b
  const int T_IN_W = 15, T_IN_B = 16;
  const int TL0 = 17;                 // 12 tensors per transformer layer
  const int T_OUT_W = 65, T_OUT_B = 66;
  const int F_SP_W = 67, F_SP_B = 68, F_TP_W = 69, F_TP_B = 70;
  const int MHA_IN_W = 71, MHA_IN_B = 72, MHA_OUT_W = 73, MHA_OUT_B = 74;
  const int FUS_W = 75, FUS_B = 76, O1_W = 77, O1_B = 78, O2_W = 79, O2_B = 80;
  const int H1_W = 81, H1_B = 82, H2_W = 83, H2_B = 84;

  // workspace arenas (floats)
  float* ws = (float*)d_ws;
  size_t NH  = (size_t)N_NODES * HID;        // 6.4M
  size_t NHh = (size_t)N_NODES * HEADS;      // 0.4M
  float* xs   = ws;                          // spatial activations (persist)
  float* hbuf = xs + NH;                     // GAT h; later reused as qp
  float* agg  = hbuf + NH;                   // GAT agg; later reused as sp_proj
  float* a_s  = agg + NH;
  float* a_d  = a_s + NHh;
  float* mmax = a_d + NHh;
  float* ssum = mmax + NHh;
  float* exb  = ssum + NHh;                  // E*HEADS = 6.4M
  // temporal scratch reuses exb arena (runs after GAT completes)
  float* t_buf  = exb;
  float* t_tmp  = t_buf  + BATCH * SEQ * HID;
  float* qkvb   = t_tmp  + BATCH * SEQ * HID;
  float* attn_t = qkvb   + BATCH * SEQ * 3 * HID;
  float* ffb    = attn_t + BATCH * SEQ * HID;
  float* tmp2   = ffb    + BATCH * SEQ * 4 * HID;
  float* kp     = tmp2   + BATCH * SEQ * HID;
  float* vp     = kp     + BATCH * SEQ * HID;
  float* spp = agg;                          // reuse after GAT
  float* qp  = hbuf;                         // reuse after GAT
  float* BN1 = exb + (size_t)N_EDGES * HEADS;           // [B*N,128]
  float* BN2 = BN1 + (size_t)BATCH * N_NODES * HID;     // [B*N,128]
  _Float16* wpack = (_Float16*)(BN2 + (size_t)BATCH * N_NODES * HID); // 64K f16 max
  // padded-K staging (used before BN1's real use in the fusion stage)
  float* nf_pad = BN1;                       // [50000, 32]
  float* ts_pad = BN1 + (size_t)N_NODES * 32;// [384, 32]

  auto gemm = [&](const float* A, const float* B, const float* bias, float* C,
                  int M, int N, int Korig, int ldb, int act) {
    int Kpad = (Korig + 31) & ~31;
    int nb = N * Kpad;
    pack_b_kernel<<<(nb + 255) / 256, 256, 0, stream>>>(B, wpack, Korig, Kpad, N, ldb);
    dim3 g((unsigned)((M / 16 + 3) / 4), (unsigned)(N / 32));
    gemm_wmma<<<g, 128, 0, stream>>>(A, wpack, bias, C, M, N, Kpad, act);
  };
  auto fill = [&](float* p, float v, int n) {
    fill_kernel<<<(n + 255) / 256, 256, 0, stream>>>(p, v, n);
  };

  // ---- Spatial encoder: input proj + 2 residual GAT layers ----
  pad_a_kernel<<<(N_NODES * 32 + 255) / 256, 256, 0, stream>>>(nf, nf_pad, FEAT, 32,
                                                               N_NODES * 32);
  gemm(nf_pad, P(SP_IN_W), P(SP_IN_B), xs, N_NODES, HID, FEAT, HID, 0);
  const int NH8 = N_NODES * HEADS;
  const int EH  = N_EDGES * HEADS;
  for (int l = 0; l < 2; ++l) {
    int base = GAT0 + l * 5;
    gemm(xs, P(base + 0), nullptr, hbuf, N_NODES, HID, HID, HID, 0);
    gat_alpha_kernel<<<(NH8 + 255) / 256, 256, 0, stream>>>(hbuf, P(base + 1), P(base + 2),
                                                            a_s, a_d, NH8);
    fill(mmax, -1e30f, NH8);
    fill(ssum, 0.0f, NH8);
    fill(agg, 0.0f, (int)NH);
    edge_max_kernel<<<(EH + 255) / 256, 256, 0, stream>>>(src, dst, a_s, a_d, mmax, EH);
    edge_expsum_kernel<<<(EH + 255) / 256, 256, 0, stream>>>(src, dst, a_s, a_d, mmax,
                                                             exb, ssum, EH);
    edge_agg_kernel<<<(EH + 255) / 256, 256, 0, stream>>>(src, dst, exb, ssum, hbuf, agg, EH);
    residual_ln_kernel<<<N_NODES, HID, 0, stream>>>(agg, xs, P(base + 3), P(base + 4), xs, 1);
  }

  // ---- Temporal encoder ----
  const int BS = BATCH * SEQ;                // 384 tokens
  pad_a_kernel<<<(BS * 32 + 255) / 256, 256, 0, stream>>>(ts, ts_pad, FEAT, 32, BS * 32);
  gemm(ts_pad, P(T_IN_W), P(T_IN_B), t_buf, BS, HID, FEAT, HID, 0);
  add_pe_kernel<<<(BS * HID + 255) / 256, 256, 0, stream>>>(t_buf, BS * HID);
  for (int l = 0; l < 4; ++l) {
    int base = TL0 + l * 12;
    gemm(t_buf, P(base + 0), P(base + 1), qkvb, BS, 3 * HID, HID, 3 * HID, 0);
    temporal_attn_kernel<<<BATCH * HEADS, SEQ, 0, stream>>>(qkvb, attn_t);
    gemm(attn_t, P(base + 2), P(base + 3), t_tmp, BS, HID, HID, HID, 0);
    residual_ln_kernel<<<BS, HID, 0, stream>>>(t_tmp, t_buf, P(base + 4), P(base + 5), t_buf, 0);
    gemm(t_buf, P(base + 6), P(base + 7), ffb, BS, 4 * HID, HID, 4 * HID, 1);
    gemm(ffb, P(base + 8), P(base + 9), t_tmp, BS, HID, 4 * HID, HID, 0);
    residual_ln_kernel<<<BS, HID, 0, stream>>>(t_tmp, t_buf, P(base + 10), P(base + 11), t_buf, 0);
  }
  gemm(t_buf, P(T_OUT_W), P(T_OUT_B), tmp2, BS, HID, HID, HID, 0);

  // ---- Fusion: projections, cross attention, pointwise stack ----
  gemm(tmp2, P(F_TP_W), P(F_TP_B), t_tmp, BS, HID, HID, HID, 0);
  gemm(t_tmp, P(MHA_IN_W) + HID,     P(MHA_IN_B) + HID,     kp, BS, HID, HID, 3 * HID, 0);
  gemm(t_tmp, P(MHA_IN_W) + 2 * HID, P(MHA_IN_B) + 2 * HID, vp, BS, HID, HID, 3 * HID, 0);
  gemm(xs, P(F_SP_W), P(F_SP_B), spp, N_NODES, HID, HID, HID, 0);
  gemm(spp, P(MHA_IN_W), P(MHA_IN_B), qp, N_NODES, HID, HID, 3 * HID, 0);

  dim3 fg((unsigned)((N_NODES / 16 + 3) / 4), (unsigned)(BATCH * HEADS));
  fusion_attn_kernel<<<fg, 128, 0, stream>>>(qp, kp, vp, BN1);

  const int BN = BATCH * N_NODES;            // 200000 rows
  gemm(BN1, P(MHA_OUT_W), P(MHA_OUT_B), BN2, BN, HID, HID, HID, 0);
  gemm(BN2, P(FUS_W), P(FUS_B), BN1, BN, HID, HID, HID, 1);
  gemm(BN1, P(O1_W), P(O1_B), BN2, BN, 64, HID, 64, 1);
  head_kernel<<<(BN + 255) / 256, 256, 0, stream>>>(BN2, P(O2_W), P(O2_B),
                                                    P(H1_W), P(H1_B),
                                                    P(H2_W), P(H2_B),
                                                    (float*)d_out, BN);
}
